// ElasticMHA_78340203479406
// MI455X (gfx1250) — matrix operand order repaired and checked by reading, not compiled, not run on hardware
//
#include <hip/hip_runtime.h>
#include <hip/hip_bf16.h>

typedef __bf16 bf16;
typedef __attribute__((ext_vector_type(16))) __bf16 v16bf;
typedef __attribute__((ext_vector_type(8)))  float  v8f;

#define NHEADS 12
#define BATCH  32
#define SEQ    577
#define DMODEL 768
#define NTOK   (BATCH * SEQ)      // 18464
#define LQPAD  640                // 10 * 64, covers i-row tiles
#define LKPAD  608                // 19 * 32, covers j-col tiles
#define NJT    19                 // j tiles of 32
#define NKIT   24                 // GEMM k tiles of 32
#define ATT_SCALE 0.125f          // 64^-0.5

union Frag16 { uint4 u4[2]; v16bf v; bf16 b[16]; };
union FragC  { v8f v; float f[8]; };

static __device__ __forceinline__ v8f wmma_bf16(const v16bf& a, const v16bf& b, const v8f& c) {
    return __builtin_amdgcn_wmma_f32_16x16x32_bf16(false, a, false, b, (short)0, c, false, false);
}

// CDNA5 async global->LDS copy, 16 bytes per lane, tracked by ASYNCcnt.
static __device__ __forceinline__ void async_b128(unsigned lds_off, unsigned gbyte_off, const void* sbase) {
    asm volatile("global_load_async_to_lds_b128 %0, %1, %2"
                 :: "v"(lds_off), "v"(gbyte_off), "s"(sbase)
                 : "memory");
}
static __device__ __forceinline__ void wait_async0() {
    asm volatile("s_wait_asynccnt 0x0" ::: "memory");
}
static __device__ __forceinline__ unsigned lds_addr(const void* p) {
    return (unsigned)(uintptr_t)p;   // flat shared addr: low 32 bits are the LDS byte offset
}

// ---------------------------------------------------------------- cast f32 -> bf16
__global__ void cast_f32_bf16(const float* __restrict__ in, bf16* __restrict__ out, size_t n) {
    size_t i = (size_t)blockIdx.x * blockDim.x + threadIdx.x;
    size_t stride = (size_t)gridDim.x * blockDim.x;
    for (; i < n; i += stride) out[i] = (bf16)in[i];
}

// ---------------------------------------------------------------- transpose + cast weights: [K][N] f32 -> [N][K] bf16
__global__ __launch_bounds__(256)
void transpose_cast(const float* __restrict__ in, bf16* __restrict__ out) {
    __shared__ float tile[32][33];
    const int bx = blockIdx.x * 32, by = blockIdx.y * 32;   // bx: N, by: K
    const int tx = threadIdx.x & 31, ty = threadIdx.x >> 5; // 8 row-groups
    #pragma unroll
    for (int r = ty; r < 32; r += 8)
        tile[r][tx] = in[(size_t)(by + r) * DMODEL + bx + tx];
    __syncthreads();
    #pragma unroll
    for (int r = ty; r < 32; r += 8)
        out[(size_t)(bx + r) * DMODEL + by + tx] = (bf16)tile[tx][r];
}

// ---------------------------------------------------------------- relative index tables (u8)
__global__ void build_idx(unsigned char* __restrict__ ivt, unsigned char* __restrict__ iht) {
    int idx = blockIdx.x * blockDim.x + threadIdx.x;
    if (idx >= SEQ * SEQ) return;
    int i = idx / SEQ, j = idx % SEQ;
    unsigned char a = 0, c = 0;
    if (i > 0 && j > 0) {
        int r = i - 1, cc = j - 1;
        int dv = cc / 24 - r / 24;
        int dh = cc % 24 - r % 24;
        dv = min(14, max(-14, dv)); dh = min(14, max(-14, dh));
        a = (unsigned char)(dv + 15); c = (unsigned char)(dh + 15);
    }
    ivt[idx] = a; iht[idx] = c;
}

// ---------------------------------------------------------------- WMMA GEMM [M,768] x Wt[768,768]^T, async double-buffered
// mode 0: -> Qh  [bh][LQPAD][64] bf16
// mode 1: -> Kh  [bh][LKPAD][64] bf16 (natural: score-GEMM B operand)
// mode 2: -> VhT [bh][64][LKPAD] bf16 (transposed: PV-GEMM B operand)
// mode 3: -> f32 out + bias + residual
__global__ __launch_bounds__(256)
void gemm_wmma(const bf16* __restrict__ A, const bf16* __restrict__ WT,
               const float* __restrict__ bias, const float* __restrict__ resid,
               bf16* __restrict__ obf, float* __restrict__ of32, int mode, int Mrows)
{
    __shared__ __align__(16) bf16 a_s[2][64 * 32];    // [row][k]
    __shared__ __align__(16) bf16 wT_s[2][128 * 32];  // [ncol][k]

    const int tid  = threadIdx.x;
    const int lane = tid & 31;
    const int wave = tid >> 5;
    const int wm   = wave & 3;        // 4 waves across M
    const int wn   = wave >> 2;       // 2 waves across N
    const int hi   = lane >> 4;
    const int ln   = lane & 15;
    const int m0   = blockIdx.x * 64;
    const int n0   = blockIdx.y * 128;

    auto issueA = [&](int k0, int buf) {
        int row = tid >> 2, cg = (tid & 3) * 8;           // 256 chunks of 16B
        int gm = m0 + row;
        if (gm < Mrows)
            async_b128(lds_addr(&a_s[buf][row * 32 + cg]),
                       (unsigned)(((size_t)gm * DMODEL + k0 + cg) * 2), A);
    };
    auto issueW = [&](int k0, int buf) {
        #pragma unroll
        for (int s = 0; s < 2; s++) {
            int c = tid + 256 * s;                        // 512 chunks of 16B
            int ncol = c >> 1, cg = (c & 1) * 16;
            async_b128(lds_addr(&wT_s[buf][ncol * 32 + cg]),
                       (unsigned)(((size_t)(n0 + ncol) * DMODEL + k0 + cg) * 2), WT);
        }
    };

    v8f acc[4];
    #pragma unroll
    for (int t = 0; t < 4; t++) acc[t] = {};

    issueA(0, 0); issueW(0, 0);
    for (int it = 0; it < NKIT; ++it) {
        const int cur = it & 1;
        wait_async0();
        __syncthreads();
        if (it + 1 < NKIT) { issueA((it + 1) * 32, 1 - cur); issueW((it + 1) * 32, 1 - cur); }

        Frag16 af;
        {
            int row = wm * 16 + ln, kb8 = hi * 8;
            af.u4[0] = *(const uint4*)(&a_s[cur][row * 32 + kb8]);
            af.u4[1] = *(const uint4*)(&a_s[cur][row * 32 + 16 + kb8]);
        }
        #pragma unroll
        for (int nt = 0; nt < 4; nt++) {
            Frag16 bfr;
            int ncol = wn * 64 + nt * 16 + ln, kb8 = hi * 8;
            bfr.u4[0] = *(const uint4*)(&wT_s[cur][ncol * 32 + kb8]);
            bfr.u4[1] = *(const uint4*)(&wT_s[cur][ncol * 32 + 16 + kb8]);
            acc[nt] = wmma_bf16(af.v, bfr.v, acc[nt]);
        }
    }

    // epilogue
    #pragma unroll
    for (int nt = 0; nt < 4; nt++) {
        FragC c; c.v = acc[nt];
        int gn = n0 + wn * 64 + nt * 16 + ln;
        #pragma unroll
        for (int e = 0; e < 8; e++) {
            int gm = m0 + wm * 16 + e + hi * 8;
            if (gm >= Mrows) continue;
            float val = c.f[e] + bias[gn];
            if (mode == 3) {
                of32[(size_t)gm * DMODEL + gn] = val + resid[(size_t)gm * DMODEL + gn];
            } else {
                int b = gm / SEQ, i = gm % SEQ;
                int h = gn >> 6, dd = gn & 63;
                size_t bh = (size_t)b * NHEADS + h;
                if      (mode == 0) obf[(bh * LQPAD + i) * 64 + dd] = (bf16)val;
                else if (mode == 1) obf[(bh * LKPAD + i) * 64 + dd] = (bf16)val;
                else                obf[(bh * 64 + dd) * LKPAD + i] = (bf16)val;
            }
        }
    }
}

// ---------------------------------------------------------------- q . table precompute -> [bh][577][32] f32
__global__ __launch_bounds__(256)
void qtab_kernel(const bf16* __restrict__ Qh, const float* __restrict__ tab_vk,
                 const float* __restrict__ tab_hk, float* __restrict__ qv, float* __restrict__ qh)
{
    __shared__ float tv[30 * 64], th[30 * 64];
    for (int t = threadIdx.x; t < 30 * 64; t += 256) { tv[t] = tab_vk[t]; th[t] = tab_hk[t]; }
    __syncthreads();
    int idx = blockIdx.x * 256 + threadIdx.x;
    if (idx >= BATCH * NHEADS * SEQ) return;
    int bh = idx / SEQ, i = idx % SEQ;
    const bf16* src = Qh + ((size_t)bh * LQPAD + i) * 64;
    float qrow[64];
    #pragma unroll
    for (int d = 0; d < 64; d++) qrow[d] = (float)src[d];
    float* ov = qv + (size_t)idx * 32;
    float* oh = qh + (size_t)idx * 32;
    for (int t = 0; t < 30; t++) {
        float av = 0.f, ah = 0.f;
        #pragma unroll
        for (int d = 0; d < 64; d++) { av += qrow[d] * tv[t * 64 + d]; ah += qrow[d] * th[t * 64 + d]; }
        ov[t] = av; oh[t] = ah;
    }
    ov[30] = 0.f; ov[31] = 0.f; oh[30] = 0.f; oh[31] = 0.f;
}

// ---------------------------------------------------------------- fused flash attention + rel-pos
// grid: (10 i-tiles of 64, B*H). block: 128 (4 waves, one 16-row strip each)
__global__ __launch_bounds__(128)
void attn_kernel(const bf16* __restrict__ Qh, const bf16* __restrict__ Kh, const bf16* __restrict__ VhT,
                 const float* __restrict__ qvtab, const float* __restrict__ qhtab,
                 const unsigned char* __restrict__ ivg, const unsigned char* __restrict__ ihg,
                 const float* __restrict__ tab_vv, const float* __restrict__ tab_hv,
                 bf16* __restrict__ aout)
{
    // K/V ping-pong buffers overlapped with the finale table block
    __shared__ __align__(16) union Ovl {
        struct { bf16 kn[2][32 * 64]; bf16 vt[2][64 * 32]; } kv;  // kn: [j][dd], vt: [dd][j]
        bf16 tabcat[64 * 64];                                     // [dd][kk]
    } ovl;
    __shared__ float qv_s[64 * 32], qh_s[64 * 32];       // per-row q.tab dots
    __shared__ float sv_s[64 * 32], sh_s[64 * 32];       // prob bucket sums
    __shared__ unsigned char iv_s[2][64 * 32], ih_s[2][64 * 32];
    __shared__ __align__(16) bf16 pw_s[4][16 * 32];      // per-wave P staging (bf16)

    const int tid = threadIdx.x, lane = tid & 31, wave = tid >> 5;
    const int hi = lane >> 4, ln = lane & 15;
    const int bh = blockIdx.y;
    const int b = bh / NHEADS, h = bh % NHEADS;
    const int i0 = blockIdx.x * 64;

    const bf16* Kb  = Kh  + (size_t)bh * LKPAD * 64;
    const bf16* Vtb = VhT + (size_t)bh * 64 * LKPAD;

    auto issueK = [&](int j0, int buf) {
        #pragma unroll
        for (int s = 0; s < 2; s++) {
            int c = tid + 128 * s;                       // 256 chunks of 16B (32 rows x 128B)
            int jj = c >> 3, cg = (c & 7) * 8;
            async_b128(lds_addr(&ovl.kv.kn[buf][jj * 64 + cg]),
                       (unsigned)(((j0 + jj) * 64 + cg) * 2), Kb);
        }
    };
    auto issueV = [&](int j0, int buf) {
        #pragma unroll
        for (int s = 0; s < 2; s++) {
            int c = tid + 128 * s;                       // 256 chunks (64 rows x 64B)
            int dd = c >> 2, cg = (c & 3) * 8;
            async_b128(lds_addr(&ovl.kv.vt[buf][dd * 32 + cg]),
                       (unsigned)(((size_t)dd * LKPAD + j0 + cg) * 2), Vtb);
        }
    };
    auto loadIdx = [&](int j0, int buf) {
        int r = tid >> 1, cg = (tid & 1) * 16;
        int i = i0 + r;
        #pragma unroll
        for (int e = 0; e < 16; e++) {
            int j = j0 + cg + e;
            unsigned char a = 0, c2 = 0;
            if (i < SEQ && j < SEQ) { size_t o = (size_t)i * SEQ + j; a = ivg[o]; c2 = ihg[o]; }
            iv_s[buf][r * 32 + cg + e] = a; ih_s[buf][r * 32 + cg + e] = c2;
        }
    };

    for (int t = tid; t < 64 * 32; t += 128) {
        int r = t >> 5, i = i0 + r;
        float a = 0.f, c = 0.f;
        if (i < SEQ) {
            size_t o = ((size_t)bh * SEQ + i) * 32 + (t & 31);
            a = qvtab[o]; c = qhtab[o];
        }
        qv_s[t] = a; qh_s[t] = c; sv_s[t] = 0.f; sh_s[t] = 0.f;
    }

    // Q fragments for this wave's 16 rows (K = 0..31, 32..63)
    Frag16 aq0, aq1;
    {
        int row = i0 + wave * 16 + ln;
        const bf16* qp = Qh + ((size_t)bh * LQPAD + row) * 64;
        int kb8 = hi * 8;
        aq0.u4[0] = *(const uint4*)(qp + kb8);       aq0.u4[1] = *(const uint4*)(qp + 16 + kb8);
        aq1.u4[0] = *(const uint4*)(qp + 32 + kb8);  aq1.u4[1] = *(const uint4*)(qp + 48 + kb8);
    }

    float m[8], l[8];
    #pragma unroll
    for (int e = 0; e < 8; e++) { m[e] = -1e30f; l[e] = 0.f; }

    // ---------------- PASS 1: online row max + row sum
    issueK(0, 0); loadIdx(0, 0);
    for (int it = 0; it < NJT; ++it) {
        const int j0 = it * 32, cur = it & 1;
        wait_async0();
        __syncthreads();
        if (it + 1 < NJT) { issueK(j0 + 32, 1 - cur); loadIdx(j0 + 32, 1 - cur); }

        float s[2][8];
        #pragma unroll
        for (int nt = 0; nt < 2; nt++) {
            v8f sc = {};
            #pragma unroll
            for (int ks = 0; ks < 2; ks++) {
                Frag16 bfr;
                int ncol = nt * 16 + ln, off = ks * 32 + hi * 8;
                bfr.u4[0] = *(const uint4*)(&ovl.kv.kn[cur][ncol * 64 + off]);
                bfr.u4[1] = *(const uint4*)(&ovl.kv.kn[cur][ncol * 64 + off + 16]);
                sc = wmma_bf16(ks ? aq1.v : aq0.v, bfr.v, sc);
            }
            FragC c; c.v = sc;
            int j = j0 + nt * 16 + ln;
            #pragma unroll
            for (int e = 0; e < 8; e++) {
                int r = wave * 16 + e + hi * 8;
                float val = -1e30f;
                if (j < SEQ) {
                    int t1 = iv_s[cur][r * 32 + (j - j0)];
                    int t2 = ih_s[cur][r * 32 + (j - j0)];
                    val = c.f[e] * ATT_SCALE + qv_s[r * 32 + t1] + qh_s[r * 32 + t2];
                }
                s[nt][e] = val;
            }
        }
        #pragma unroll
        for (int e = 0; e < 8; e++) {
            float tm = fmaxf(s[0][e], s[1][e]);
            tm = fmaxf(tm, __shfl_xor(tm, 1, 32));
            tm = fmaxf(tm, __shfl_xor(tm, 2, 32));
            tm = fmaxf(tm, __shfl_xor(tm, 4, 32));
            tm = fmaxf(tm, __shfl_xor(tm, 8, 32));
            float mn = fmaxf(m[e], tm);
            float part = __expf(s[0][e] - mn) + __expf(s[1][e] - mn);
            part += __shfl_xor(part, 1, 32);
            part += __shfl_xor(part, 2, 32);
            part += __shfl_xor(part, 4, 32);
            part += __shfl_xor(part, 8, 32);
            l[e] = l[e] * __expf(m[e] - mn) + part;
            m[e] = mn;
        }
    }
    __syncthreads();

    // ---------------- PASS 2: P, O += P@V, bucket sums
    v8f o[4];
    #pragma unroll
    for (int nt = 0; nt < 4; nt++) o[nt] = {};

    issueK(0, 0); issueV(0, 0); loadIdx(0, 0);
    for (int it = 0; it < NJT; ++it) {
        const int j0 = it * 32, cur = it & 1;
        wait_async0();
        __syncthreads();
        if (it + 1 < NJT) { issueK(j0 + 32, 1 - cur); issueV(j0 + 32, 1 - cur); loadIdx(j0 + 32, 1 - cur); }

        #pragma unroll
        for (int nt = 0; nt < 2; nt++) {
            v8f sc = {};
            #pragma unroll
            for (int ks = 0; ks < 2; ks++) {
                Frag16 bfr;
                int ncol = nt * 16 + ln, off = ks * 32 + hi * 8;
                bfr.u4[0] = *(const uint4*)(&ovl.kv.kn[cur][ncol * 64 + off]);
                bfr.u4[1] = *(const uint4*)(&ovl.kv.kn[cur][ncol * 64 + off + 16]);
                sc = wmma_bf16(ks ? aq1.v : aq0.v, bfr.v, sc);
            }
            FragC c; c.v = sc;
            int j = j0 + nt * 16 + ln;
            #pragma unroll
            for (int e = 0; e < 8; e++) {
                int r = wave * 16 + e + hi * 8;
                float p = 0.f;
                if (j < SEQ) {
                    int t1 = iv_s[cur][r * 32 + (j - j0)];
                    int t2 = ih_s[cur][r * 32 + (j - j0)];
                    float val = c.f[e] * ATT_SCALE + qv_s[r * 32 + t1] + qh_s[r * 32 + t2];
                    p = __expf(val - m[e]);
                    atomicAdd(&sv_s[r * 32 + t1], p);
                    atomicAdd(&sh_s[r * 32 + t2], p);
                }
                pw_s[wave][(e + hi * 8) * 32 + nt * 16 + ln] = (bf16)p;
            }
        }
        asm volatile("s_wait_dscnt 0x0" ::: "memory");   // wave-local LDS RAW on pw_s
        __builtin_amdgcn_wave_barrier();

        Frag16 pf;   // P fragment, rows of this wave's tile
        {
            int kb8 = hi * 8;
            const bf16* pr = &pw_s[wave][ln * 32];
            pf.u4[0] = *(const uint4*)(pr + kb8);
            pf.u4[1] = *(const uint4*)(pr + 16 + kb8);
        }
        #pragma unroll
        for (int nt = 0; nt < 4; nt++) {
            Frag16 bfr;
            int dcol = nt * 16 + ln, kb8 = hi * 8;
            bfr.u4[0] = *(const uint4*)(&ovl.kv.vt[cur][dcol * 32 + kb8]);
            bfr.u4[1] = *(const uint4*)(&ovl.kv.vt[cur][dcol * 32 + 16 + kb8]);
            o[nt] = wmma_bf16(pf.v, bfr.v, o[nt]);
        }
    }
    __syncthreads();   // all K/V reads done before tabcat overlays the union

    // ---------------- finale: O += buckets @ [tab_vv; tab_hv], normalize, store
    for (int t = tid; t < 64 * 64; t += 128) {
        int dd = t >> 6, kk = t & 63;
        float val = 0.f;
        if (kk < 30)                  val = tab_vv[kk * 64 + dd];
        else if (kk >= 32 && kk < 62) val = tab_hv[(kk - 32) * 64 + dd];
        ovl.tabcat[dd * 64 + kk] = (bf16)val;
    }
    __syncthreads();

    Frag16 s0, s1;
    {
        int row = wave * 16 + ln, kb8 = hi * 8;
        const float* pv = &sv_s[row * 32];
        const float* ph = &sh_s[row * 32];
        #pragma unroll
        for (int e = 0; e < 8; e++) { s0.v[e] = (bf16)pv[kb8 + e];          s1.v[e] = (bf16)ph[kb8 + e]; }
        #pragma unroll
        for (int e = 0; e < 8; e++) { s0.v[8 + e] = (bf16)pv[16 + kb8 + e]; s1.v[8 + e] = (bf16)ph[16 + kb8 + e]; }
    }
    #pragma unroll
    for (int nt = 0; nt < 4; nt++) {
        Frag16 b0, b1;
        int dcol = nt * 16 + ln, kb8 = hi * 8;
        b0.u4[0] = *(const uint4*)(&ovl.tabcat[dcol * 64 + kb8]);
        b0.u4[1] = *(const uint4*)(&ovl.tabcat[dcol * 64 + 16 + kb8]);
        b1.u4[0] = *(const uint4*)(&ovl.tabcat[dcol * 64 + 32 + kb8]);
        b1.u4[1] = *(const uint4*)(&ovl.tabcat[dcol * 64 + 48 + kb8]);
        o[nt] = wmma_bf16(s0.v, b0.v, o[nt]);
        o[nt] = wmma_bf16(s1.v, b1.v, o[nt]);
    }
    #pragma unroll
    for (int nt = 0; nt < 4; nt++) {
        FragC c; c.v = o[nt];
        int dd = nt * 16 + ln;
        #pragma unroll
        for (int e = 0; e < 8; e++) {
            int i = i0 + wave * 16 + e + hi * 8;
            if (i >= SEQ) continue;
            float val = c.f[e] / l[e];
            aout[((size_t)b * SEQ + i) * DMODEL + h * 64 + dd] = (bf16)val;
        }
    }
}

// ----------------------------------------------------------------------------------
extern "C" void kernel_launch(void* const* d_in, const int* in_sizes, int n_in,
                              void* d_out, int out_size, void* d_ws, size_t ws_size,
                              hipStream_t stream) {
    const float* q      = (const float*)d_in[0];
    const float* k      = (const float*)d_in[1];
    const float* v      = (const float*)d_in[2];
    const float* Wq     = (const float*)d_in[3];
    const float* bq     = (const float*)d_in[4];
    const float* Wk     = (const float*)d_in[5];
    const float* bk     = (const float*)d_in[6];
    const float* Wv     = (const float*)d_in[7];
    const float* bv     = (const float*)d_in[8];
    const float* Wp     = (const float*)d_in[9];
    const float* bp     = (const float*)d_in[10];
    const float* tab_vk = (const float*)d_in[11];
    const float* tab_hk = (const float*)d_in[12];
    const float* tab_vv = (const float*)d_in[13];
    const float* tab_hv = (const float*)d_in[14];
    float* out = (float*)d_out;

    char* ws = (char*)d_ws;
    size_t off = 0;
    auto alloc = [&](size_t bytes) -> char* {
        char* p = ws + off;
        off += (bytes + 255) & ~(size_t)255;
        return p;
    };

    const size_t act_elems = (size_t)NTOK * DMODEL;
    bf16* qbf = (bf16*)alloc(act_elems * 2);
    bf16* kbf = (bf16*)alloc(act_elems * 2);
    bf16* vbf = (bf16*)alloc(act_elems * 2);
    bf16* wqt = (bf16*)alloc((size_t)DMODEL * DMODEL * 2);  // transposed bf16 weights
    bf16* wkt = (bf16*)alloc((size_t)DMODEL * DMODEL * 2);
    bf16* wvt = (bf16*)alloc((size_t)DMODEL * DMODEL * 2);
    bf16* wpt = (bf16*)alloc((size_t)DMODEL * DMODEL * 2);
    const size_t BH = (size_t)BATCH * NHEADS;
    bf16* Qh  = (bf16*)alloc(BH * LQPAD * 64 * 2);
    bf16* Khn = (bf16*)alloc(BH * LKPAD * 64 * 2);
    bf16* VhT = (bf16*)alloc(BH * 64 * LKPAD * 2);
    float* qvt = (float*)alloc(BH * SEQ * 32 * 4);
    float* qht = (float*)alloc(BH * SEQ * 32 * 4);
    unsigned char* ivb = (unsigned char*)alloc((size_t)SEQ * SEQ);
    unsigned char* ihb = (unsigned char*)alloc((size_t)SEQ * SEQ);
    bf16* aout = (bf16*)alloc(act_elems * 2);

    // 1) casts + weight transposes
    cast_f32_bf16<<<4096, 256, 0, stream>>>(q, qbf, act_elems);
    cast_f32_bf16<<<4096, 256, 0, stream>>>(k, kbf, act_elems);
    cast_f32_bf16<<<4096, 256, 0, stream>>>(v, vbf, act_elems);
    dim3 gt(DMODEL / 32, DMODEL / 32);
    transpose_cast<<<gt, 256, 0, stream>>>(Wq, wqt);
    transpose_cast<<<gt, 256, 0, stream>>>(Wk, wkt);
    transpose_cast<<<gt, 256, 0, stream>>>(Wv, wvt);
    transpose_cast<<<gt, 256, 0, stream>>>(Wp, wpt);

    // 2) index tables
    build_idx<<<(SEQ * SEQ + 255) / 256, 256, 0, stream>>>(ivb, ihb);

    // 3) QKV projections (WMMA, async double-buffered)
    dim3 gg((NTOK + 63) / 64, DMODEL / 128);
    gemm_wmma<<<gg, 256, 0, stream>>>(qbf, wqt, bq, nullptr, Qh, nullptr, 0, NTOK);
    gemm_wmma<<<gg, 256, 0, stream>>>(kbf, wkt, bk, nullptr, Khn, nullptr, 1, NTOK);
    gemm_wmma<<<gg, 256, 0, stream>>>(vbf, wvt, bv, nullptr, VhT, nullptr, 2, NTOK);

    // 4) q . rel-table dots
    qtab_kernel<<<((int)(BH * SEQ) + 255) / 256, 256, 0, stream>>>(Qh, tab_vk, tab_hk, qvt, qht);

    // 5) fused flash attention + relative bias + bucketed value-side term
    dim3 ga((SEQ + 63) / 64, (unsigned)BH);
    attn_kernel<<<ga, 128, 0, stream>>>(Qh, Khn, VhT, qvt, qht, ivb, ihb, tab_vv, tab_hv, aout);

    // 6) output projection + bias + residual (f32 out)
    gemm_wmma<<<gg, 256, 0, stream>>>(aout, wpt, bp, q, nullptr, out, 3, NTOK);
}
